// GraphVAE_5815385719161
// MI455X (gfx1250) — compile-verified, matrix operand order checked by reading, not run on hardware
//
#include <hip/hip_runtime.h>
#include <hip/hip_bf16.h>

// ---------------------------------------------------------------------------
// GraphVAE forward for MI455X (gfx1250, wave32).
// Memory-bound workload: all GEMMs via v_wmma_f32_16x16x32_f16 with
//  - B fragments pre-packed to f16 WMMA layout in LDS (1 b128-pair per frag)
//  - A fragments loaded as float4 pairs (contiguous K runs of the layout)
//  - GN+GELU fused into the A-fragment load, GN tables staged in LDS
// Graph conv via native f32 global atomics; GN stats via LDS ds_add_f32.
// ---------------------------------------------------------------------------

typedef __attribute__((ext_vector_type(16))) _Float16 v16h;
typedef __attribute__((ext_vector_type(8)))  float    v8f;

#define N_ET 7
#define N_BATCH 8

__device__ __forceinline__ float gelu_f(float x) {
    const float c0 = 0.7978845608028654f;   // sqrt(2/pi)
    float x3 = x * x * x;
    return 0.5f * x * (1.0f + tanhf(c0 * (x + 0.044715f * x3)));
}

// ---------------------------------------------------------------------------
__global__ void k_zero(float* __restrict__ p, long n) {
    long i = (long)blockIdx.x * blockDim.x + threadIdx.x;
    if (i < n) p[i] = 0.0f;
}

// ---------------------------------------------------------------------------
// per-edge gather + typed scatter-add:  agg[dst][et*CIN + c] += x[src][c]
template <int CIN>
__global__ void k_scatter(const float* __restrict__ x,
                          const int* __restrict__ src,
                          const int* __restrict__ dst,
                          const int* __restrict__ et,
                          float* __restrict__ agg, int E) {
    int e = blockIdx.x * blockDim.x + threadIdx.x;
    if (e >= E) return;
    int s = src[e], d = dst[e], t = et[e];
    const float4* xp4 = (const float4*)(x + (long)s * CIN);
    float vals[CIN];
    #pragma unroll
    for (int q = 0; q < CIN / 4; ++q) {
        float4 f = xp4[q];
        vals[4 * q + 0] = f.x; vals[4 * q + 1] = f.y;
        vals[4 * q + 2] = f.z; vals[4 * q + 3] = f.w;
    }
    float* ap = agg + (long)d * (N_ET * CIN) + t * CIN;
    #pragma unroll
    for (int c = 0; c < CIN; ++c)
        unsafeAtomicAdd(&ap[c], vals[c]);   // global_atomic_add_f32
}

// ---------------------------------------------------------------------------
// GN stats: per-(batch,channel) sum / sum^2 / count (gs==1 for all GNs here).
// Block-local LDS reduction (ds_add_f32) then one global atomic per slot.
__global__ void k_gn_stats(const float* __restrict__ x,
                           const int* __restrict__ bid, int N, int C,
                           float* __restrict__ s1, float* __restrict__ s2,
                           float* __restrict__ cnt) {
    __shared__ float ls1[N_BATCH * 32];
    __shared__ float ls2[N_BATCH * 32];
    __shared__ float lc[N_BATCH];
    int tid = threadIdx.x;
    for (int i = tid; i < N_BATCH * 32; i += blockDim.x) { ls1[i] = 0.f; ls2[i] = 0.f; }
    if (tid < N_BATCH) lc[tid] = 0.f;
    __syncthreads();
    int n = blockIdx.x * blockDim.x + tid;
    if (n < N) {
        int b = bid[n];
        atomicAdd(&lc[b], 1.0f);
        const float4* xp4 = (const float4*)(x + (long)n * C);
        for (int q = 0; q < (C >> 2); ++q) {
            float4 f = xp4[q];
            int c = q << 2;
            atomicAdd(&ls1[b * C + c + 0], f.x); atomicAdd(&ls2[b * C + c + 0], f.x * f.x);
            atomicAdd(&ls1[b * C + c + 1], f.y); atomicAdd(&ls2[b * C + c + 1], f.y * f.y);
            atomicAdd(&ls1[b * C + c + 2], f.z); atomicAdd(&ls2[b * C + c + 2], f.z * f.z);
            atomicAdd(&ls1[b * C + c + 3], f.w); atomicAdd(&ls2[b * C + c + 3], f.w * f.w);
        }
    }
    __syncthreads();
    for (int i = tid; i < N_BATCH * C; i += blockDim.x) {
        if (ls1[i] != 0.f) unsafeAtomicAdd(&s1[i], ls1[i]);
        if (ls2[i] != 0.f) unsafeAtomicAdd(&s2[i], ls2[i]);
    }
    if (tid < N_BATCH && lc[tid] != 0.f) unsafeAtomicAdd(&cnt[tid], lc[tid]);
}

__global__ void k_gn_finalize(const float* __restrict__ s1, const float* __restrict__ s2,
                              const float* __restrict__ cnt,
                              float* __restrict__ mean, float* __restrict__ inv, int C) {
    int i = threadIdx.x;
    if (i >= N_BATCH * C) return;
    int b = i / C;
    float cn  = fmaxf(cnt[b], 1.0f);
    float m   = s1[i] / cn;
    float var = s2[i] / cn - m * m;
    mean[i] = m;
    inv[i]  = rsqrtf(var + 1e-5f);
}

// elementwise GN + GELU (only needed before the edge gather)
__global__ void k_gn_gelu(const float* __restrict__ x, const int* __restrict__ bid,
                          const float* __restrict__ mean, const float* __restrict__ inv,
                          const float* __restrict__ gamma, const float* __restrict__ beta,
                          float* __restrict__ y, int N, int C) {
    long i = (long)blockIdx.x * blockDim.x + threadIdx.x;
    if (i >= (long)N * C) return;
    int n = (int)(i / C), c = (int)(i % C);
    int b = bid[n];
    float v = (x[i] - mean[b * C + c]) * inv[b * C + c] * gamma[c] + beta[c];
    y[i] = gelu_f(v);
}

// ---------------------------------------------------------------------------
// WMMA GEMM:  Out[., col0 + 0..Nc) = scale * (A' @ B) + bias + residual
//   A  : [M,K] f32 row-major; if bid != null, A' = gelu(GN(A)) fused in load
//   B  : weights, wT ? W[n*K+k] ([Nc,K] i.e. x@W.T) : W[k*Nc+n] ([K,Nc])
//   one wave32 per 16-row tile; K <= 64 (chunks of 32); Nc <= 32.
//   B fragments pre-packed to f16 WMMA layout in LDS.
// ---------------------------------------------------------------------------
#define GEMM_WAVES 4
__global__ void k_wmma_gemm(const float* __restrict__ A, int M, int K,
                            const float* __restrict__ W, int wT, int Nc,
                            const float* __restrict__ bias,
                            const float* __restrict__ residual, float scale,
                            const int* __restrict__ bid,
                            const float* __restrict__ mean, const float* __restrict__ inv,
                            const float* __restrict__ gamma, const float* __restrict__ beta,
                            float* __restrict__ Out, int ldOut, int col0) {
    // B fragments in register layout: [kChunk][colTile][lane][j]
    __shared__ _Float16 Bf[2][2][32][16];
    __shared__ float gnMean[N_BATCH * 32], gnInv[N_BATCH * 32];
    __shared__ float gnG[32], gnB[32];
    int tid = threadIdx.x;
    for (int i = tid; i < 2 * 2 * 32 * 16; i += blockDim.x) {
        int j  = i & 15;
        int l  = (i >> 4) & 31;
        int t  = (i >> 9) & 1;
        int kc = i >> 10;
        // 16-bit WMMA A/B layout: k = 32*kc + 16*(j>>3) + 8*(lane>>4) + (j&7)
        int k = (kc << 5) + ((j >> 3) << 4) + ((l >> 4) << 3) + (j & 7);
        int n = (t << 4) + (l & 15);
        float v = 0.0f;
        if (k < K && n < Nc) v = wT ? W[n * K + k] : W[k * Nc + n];
        Bf[kc][t][l][j] = (_Float16)v;
    }
    if (bid) {   // GN only used with K == 8 or 32
        for (int i = tid; i < N_BATCH * K; i += blockDim.x) {
            gnMean[i] = mean[i];
            gnInv[i]  = inv[i];
        }
        for (int i = tid; i < K; i += blockDim.x) { gnG[i] = gamma[i]; gnB[i] = beta[i]; }
    }
    __syncthreads();

    int wave = tid >> 5, lane = tid & 31;
    int m0 = (blockIdx.x * GEMM_WAVES + wave) * 16;
    if (m0 >= M) return;                  // wave-uniform: EXEC stays full for WMMA

    int halfSel = lane >> 4;
    int l15 = lane & 15;
    int aRow = m0 + l15;                  // A fragment: M = lane&15
    bool rowOk = aRow < M;
    int b = (bid && rowOk) ? bid[aRow] : 0;
    const float* Arow = A + (long)aRow * K;

    v8f acc0 = {}; v8f acc1 = {};
    int kChunks = (K + 31) >> 5;
    for (int kc = 0; kc < kChunks; ++kc) {
        int k0 = kc << 5;
        v16h af;
        #pragma unroll
        for (int run = 0; run < 2; ++run) {
            // per-lane contiguous 8-float run: k = kbase .. kbase+7
            int kbase = k0 + (run << 4) + (halfSel << 3);
            float v[8];
            if (rowOk && kbase + 8 <= K) {          // fast path: 2x b128 loads
                const float4* p = (const float4*)(Arow + kbase);
                float4 f0 = p[0], f1 = p[1];
                v[0] = f0.x; v[1] = f0.y; v[2] = f0.z; v[3] = f0.w;
                v[4] = f1.x; v[5] = f1.y; v[6] = f1.z; v[7] = f1.w;
            } else {
                #pragma unroll
                for (int t = 0; t < 8; ++t) {
                    int k = kbase + t;
                    v[t] = (rowOk && k < K) ? Arow[k] : 0.0f;
                }
            }
            if (bid) {
                #pragma unroll
                for (int t = 0; t < 8; ++t) {
                    int k = kbase + t;
                    if (k < K) {
                        float u = (v[t] - gnMean[b * K + k]) * gnInv[b * K + k]
                                  * gnG[k] + gnB[k];
                        v[t] = gelu_f(u);
                    }
                }
            }
            #pragma unroll
            for (int t = 0; t < 8; ++t) af[(run << 3) + t] = (_Float16)v[t];
        }
        v16h bf0 = *(const v16h*)(&Bf[kc][0][lane][0]);
        acc0 = __builtin_amdgcn_wmma_f32_16x16x32_f16(false, af, false, bf0,
                                                      (short)0, acc0, false, false);
        if (Nc > 16) {
            v16h bf1 = *(const v16h*)(&Bf[kc][1][lane][0]);
            acc1 = __builtin_amdgcn_wmma_f32_16x16x32_f16(false, af, false, bf1,
                                                          (short)0, acc1, false, false);
        }
    }

    // C/D layout: row = m0 + 8*(lane>>4) + i, col = lane&15 (+16 for tile 1)
    #pragma unroll
    for (int i = 0; i < 8; ++i) {
        int row = m0 + (halfSel << 3) + i;
        if (row >= M) continue;
        int c = l15;
        if (c < Nc) {
            float v = acc0[i] * scale;
            if (bias)     v += bias[c];
            if (residual) v += residual[(long)row * Nc + c];
            Out[(long)row * ldOut + col0 + c] = v;
        }
        int c2 = 16 + l15;
        if (c2 < Nc) {
            float v = acc1[i] * scale;
            if (bias)     v += bias[c2];
            if (residual) v += residual[(long)row * Nc + c2];
            Out[(long)row * ldOut + col0 + c2] = v;
        }
    }
}

// ---------------------------------------------------------------------------
// reparameterization sample: z = mean + exp(0.5*clip(logvar)) * N(0,1)
// deterministic hash-based Box-Muller (graph-capture safe)
__global__ void k_sample(const float* __restrict__ code, float* __restrict__ z, int N) {
    int i = blockIdx.x * blockDim.x + threadIdx.x;
    if (i >= N * 3) return;
    int n = i / 3, e = i % 3;
    float m  = code[n * 6 + e];
    float lv = fminf(fmaxf(code[n * 6 + 3 + e], -30.0f), 20.0f);
    unsigned s = (unsigned)i;
    unsigned h1 = s * 0x9E3779B1u; h1 ^= h1 >> 16; h1 *= 0x85EBCA6Bu; h1 ^= h1 >> 13;
    h1 *= 0xC2B2AE35u; h1 ^= h1 >> 16;
    unsigned h2 = (s ^ 0xDEADBEEFu) * 0x27D4EB2Fu; h2 ^= h2 >> 15; h2 *= 0x165667B1u; h2 ^= h2 >> 13;
    float u1 = ((h1 >> 8) + 1u) * (1.0f / 16777217.0f);
    float u2 = (h2 >> 8) * (1.0f / 16777216.0f);
    float nz = sqrtf(-2.0f * logf(u1)) * cosf(6.28318530718f * u2);
    z[n * 3 + e] = m + expf(0.5f * lv) * nz;
}

// tanh post-processing on output cols 6..9 (0.5*tanh for cols 6..8)
__global__ void k_point_post(float* __restrict__ out, int N) {
    int i = blockIdx.x * blockDim.x + threadIdx.x;
    if (i >= N * 4) return;
    int n = i / 4, j = i % 4;
    float t = tanhf(out[(long)n * 10 + 6 + j]);
    out[(long)n * 10 + 6 + j] = (j < 3) ? 0.5f * t : t;
}

// ---------------------------------------------------------------------------
extern "C" void kernel_launch(void* const* d_in, const int* in_sizes, int n_in,
                              void* d_out, int out_size, void* d_ws, size_t ws_size,
                              hipStream_t stream) {
    const int N = in_sizes[0] / 4;
    const int E = in_sizes[1];
    const float* x   = (const float*)d_in[0];
    const int*   src = (const int*)d_in[1];
    const int*   dst = (const int*)d_in[2];
    const int*   et  = (const int*)d_in[3];
    const int*   bid = (const int*)d_in[4];
    auto F = [&](int i) { return (const float*)d_in[i]; };

    // params leaves: pytree order (dict keys sorted alphabetically, lists in order)
    struct RBp { const float *Wa,*Wb,*Wg,*b1,*b2,*b3,*ba,*bb,*g1,*g2,*g3; };
    struct Hp  { const float *W1,*W2,*b,*b1,*b2,*g; };
    int p = 5;
    auto loadRB = [&](int& q) { RBp r; r.Wa=F(q++); r.Wb=F(q++); r.Wg=F(q++);
        r.b1=F(q++); r.b2=F(q++); r.b3=F(q++); r.ba=F(q++); r.bb=F(q++);
        r.g1=F(q++); r.g2=F(q++); r.g3=F(q++); return r; };
    auto loadH  = [&](int& q) { Hp h; h.W1=F(q++); h.W2=F(q++); h.b=F(q++);
        h.b1=F(q++); h.b2=F(q++); h.g=F(q++); return h; };

    const float* KL_W = F(p++); const float* KL_b = F(p++);
    const float* conv1_W = F(p++);
    RBp dec[3];  for (int i = 0; i < 3; ++i) dec[i]  = loadRB(p);
    RBp enc[2];  for (int i = 0; i < 2; ++i) enc[i]  = loadRB(p);
    const float* enc_b = F(p++); const float* enc_g = F(p++);
    RBp mid1[3]; for (int i = 0; i < 3; ++i) mid1[i] = loadRB(p);
    RBp mid2[3]; for (int i = 0; i < 3; ++i) mid2[i] = loadRB(p);
    Hp point = loadH(p);
    const float* postKL_W = F(p++); const float* postKL_b = F(p++);
    Hp pred = loadH(p);
    Hp reg  = loadH(p);

    // workspace layout (floats), ~237 MB total
    float* ws   = (float*)d_ws;
    float* hA   = ws;
    float* hB   = hA  + (long)N * 32;
    float* t1   = hB  + (long)N * 32;
    float* t2   = t1  + (long)N * 8;
    float* agg  = t2  + (long)N * 8;
    float* code = agg + (long)N * 56;
    float* zb   = code + (long)N * 8;
    float* st   = zb  + (long)N * 4;
    float* s1b = st;        float* s2b  = st + 256;  float* cntb = st + 512;
    float* mnb = st + 520;  float* invb = st + 776;

    auto zero = [&](float* pz, long n) {
        k_zero<<<dim3((unsigned)((n + 255) / 256)), 256, 0, stream>>>(pz, n);
    };
    auto gstats = [&](const float* xx, int C) {
        zero(st, 520);
        k_gn_stats<<<(N + 255) / 256, 256, 0, stream>>>(xx, bid, N, C, s1b, s2b, cntb);
        k_gn_finalize<<<1, 256, 0, stream>>>(s1b, s2b, cntb, mnb, invb, C);
    };
    auto gemm = [&](const float* A, int K, const float* W, int wT, int Nc,
                    const float* bias, const float* res, float scale, int useGN,
                    const float* gamma, const float* beta,
                    float* Out, int ldOut, int col0) {
        int tiles = (N + 15) / 16;
        int blocks = (tiles + GEMM_WAVES - 1) / GEMM_WAVES;
        k_wmma_gemm<<<blocks, 32 * GEMM_WAVES, 0, stream>>>(
            A, N, K, W, wT, Nc, bias, res, scale,
            useGN ? bid : nullptr, mnb, invb, gamma, beta, Out, ldOut, col0);
    };
    auto resblk = [&](const float* in, float* out, const RBp& r) {
        gstats(in, 32);                                                 // GN1 stats
        gemm(in, 32, r.Wa, 1, 8, r.ba, nullptr, 1.f, 1, r.g1, r.b1, t1, 8, 0);  // gelu(GN)->Wa
        gstats(t1, 8);                                                  // GN2 stats
        k_gn_gelu<<<(unsigned)(((long)N * 8 + 255) / 256), 256, 0, stream>>>(
            t1, bid, mnb, invb, r.g2, r.b2, t2, N, 8);                  // materialize for gather
        zero(agg, (long)N * 56);
        k_scatter<8><<<(E + 255) / 256, 256, 0, stream>>>(t2, src, dst, et, agg, E);
        gemm(agg, 56, r.Wg, 0, 8, nullptr, nullptr, 1.f / 7.f, 0, nullptr, nullptr, t1, 8, 0);
        gstats(t1, 8);                                                  // GN3 stats
        gemm(t1, 8, r.Wb, 1, 32, r.bb, in, 1.f, 1, r.g3, r.b3, out, 32, 0);  // + skip
    };
    auto head = [&](const float* h, float* tmp, const Hp& hp, int col0, int Nc) {
        gemm(h, 32, hp.W1, 1, 32, hp.b1, nullptr, 1.f, 0, nullptr, nullptr, tmp, 32, 0);
        gstats(tmp, 32);
        gemm(tmp, 32, hp.W2, 1, Nc, hp.b2, nullptr, 1.f, 1, hp.g, hp.b,
             (float*)d_out, 10, col0);
    };

    // ---- forward ----
    zero(agg, (long)N * 28);
    k_scatter<4><<<(E + 255) / 256, 256, 0, stream>>>(x, src, dst, et, agg, E);
    gemm(agg, 28, conv1_W, 0, 32, nullptr, nullptr, 1.f / 7.f, 0, nullptr, nullptr, hA, 32, 0);

    float* cur = hA; float* alt = hB;
    for (int i = 0; i < 2; ++i) { resblk(cur, alt, enc[i]); float* t = cur; cur = alt; alt = t; }

    gstats(cur, 32);                                                    // enc GN
    gemm(cur, 32, KL_W, 1, 6, KL_b, nullptr, 1.f, 1, enc_g, enc_b, code, 6, 0);
    k_sample<<<(N * 3 + 255) / 256, 256, 0, stream>>>(code, zb, N);
    gemm(zb, 3, postKL_W, 1, 32, postKL_b, nullptr, 1.f, 0, nullptr, nullptr, cur, 32, 0);

    for (int i = 0; i < 3; ++i) { resblk(cur, alt, mid1[i]); float* t = cur; cur = alt; alt = t; }
    for (int i = 0; i < 3; ++i) { resblk(cur, alt, mid2[i]); float* t = cur; cur = alt; alt = t; }
    for (int i = 0; i < 3; ++i) { resblk(cur, alt, dec[i]);  float* t = cur; cur = alt; alt = t; }

    head(cur, alt, pred,  0, 2);
    head(cur, alt, reg,   2, 4);
    head(cur, alt, point, 6, 4);
    k_point_post<<<(N * 4 + 255) / 256, 256, 0, stream>>>((float*)d_out, N);
}
